// MultiHeadAttention_53231824666851
// MI455X (gfx1250) — matrix operand order repaired and checked by reading, not compiled, not run on hardware
//
#include <hip/hip_runtime.h>
#include <stdint.h>

#define HEADS   16
#define D_MODEL 1024
#define DKH     64          // per-head dim
#define BATCH   2
#define SEQ     2048

typedef __attribute__((ext_vector_type(16))) __bf16   v16bf;
typedef __attribute__((ext_vector_type(8)))  float    v8f;
typedef __attribute__((ext_vector_type(4)))  uint32_t u32x4;
typedef int          i32x4v __attribute__((vector_size(16)));  // async builtin param
typedef unsigned int u32x4v __attribute__((vector_size(16)));  // TDM D# group 0
typedef int          i32x8v __attribute__((vector_size(32)));  // TDM D# group 1
typedef int          i32x4s __attribute__((vector_size(16)));  // TDM D# groups 2/3

#define AS1 __attribute__((address_space(1)))
#define AS3 __attribute__((address_space(3)))

#if defined(__has_builtin)
#  if __has_builtin(__builtin_amdgcn_global_load_async_to_lds_b128)
#    define HAVE_ASYNC_LDS 1
#  endif
#  if __has_builtin(__builtin_amdgcn_tensor_load_to_lds) && \
      __has_builtin(__builtin_amdgcn_s_wait_tensorcnt)
#    define HAVE_TDM 1
#  endif
#  if __has_builtin(__builtin_amdgcn_cvt_pk_bf16_f32)
#    define HAVE_PK_BF16 1
#  endif
#endif
#ifndef HAVE_ASYNC_LDS
#  define HAVE_ASYNC_LDS 0
#endif
#ifndef HAVE_TDM
#  define HAVE_TDM 0
#endif
#ifndef HAVE_PK_BF16
#  define HAVE_PK_BF16 0
#endif

// Fragment = 8 dwords per lane (16 bf16 elements). All members trivial.
union Frag {
    uint32_t u[8];
    u32x4    q[2];
    v16bf    v;
};

__device__ __forceinline__ uint32_t f32_to_bf16_bits(float f) {
    uint32_t u = __builtin_bit_cast(uint32_t, f);
    return (u + 0x7FFFu + ((u >> 16) & 1u)) >> 16;   // round-to-nearest-even
}
__device__ __forceinline__ uint32_t pack_bf16x2(float lo, float hi) {
#if HAVE_PK_BF16
    auto r = __builtin_amdgcn_cvt_pk_bf16_f32(lo, hi);  // one v_cvt_pk_bf16_f32
    return __builtin_bit_cast(uint32_t, r);
#else
    return f32_to_bf16_bits(lo) | (f32_to_bf16_bits(hi) << 16);
#endif
}
__device__ __forceinline__ v8f vzero8() {
    v8f z = {0.f, 0.f, 0.f, 0.f, 0.f, 0.f, 0.f, 0.f};
    return z;
}
__device__ __forceinline__ v8f wmma_bf16(v16bf a, v16bf b, v8f c) {
    return __builtin_amdgcn_wmma_f32_16x16x32_bf16(false, a, false, b, (short)0,
                                                   c, false, false);
}
__device__ __forceinline__ void stage_wait() {
#if HAVE_TDM
    __builtin_amdgcn_s_wait_tensorcnt(0);
#elif HAVE_ASYNC_LDS
    asm volatile("s_wait_asynccnt 0" ::: "memory");
#endif
}

// ---------------------------------------------------------------------------
// fp32 -> bf16 conversion
// ---------------------------------------------------------------------------
__global__ void f32_to_bf16_kernel(const float* __restrict__ src,
                                   uint16_t* __restrict__ dst, int n) {
    int i = blockIdx.x * blockDim.x + threadIdx.x;
    if (i < n) dst[i] = (uint16_t)f32_to_bf16_bits(src[i]);
}

// ---------------------------------------------------------------------------
// WMMA GEMM:  C[M,N] = A[M,K] * W[N,K]^T   (nn.Linear layout)
// OUT_MODE: 0 = bf16 row-major (scaled), 1 = f32 row-major,
//           2 = bf16 per-head transposed Vt[b][h][d][s]
// One wave owns a 64x64 C tile (4x4 fragments). All 8 waves of a block share
// one 64-row A tile, staged per-32-K-step into double-buffered LDS via the
// Tensor Data Mover (one descriptor per step, issued by wave 0), falling back
// to GLOBAL_LOAD_ASYNC_TO_LDS_B128 / plain loads. DMA overlaps the WMMAs.
// ---------------------------------------------------------------------------
#define LDS_ROW 36  // dwords per staged A row (16 data + 20 pad, keeps 16B align)

template <int OUT_MODE>
__global__ __launch_bounds__(256)
void gemm_wmma_bf16(const uint16_t* __restrict__ A,
                    const uint16_t* __restrict__ W,
                    void* __restrict__ Cout,
                    int M, int N, int Kd, float outScale) {
    __shared__ uint32_t ldsA[2][64 * LDS_ROW];

    const int tid  = threadIdx.x;
    const int lane = tid & 31;
    const int wave = blockIdx.x * (blockDim.x >> 5) + (tid >> 5);
    const int tilesN = N >> 6;
    const int tM = wave / tilesN;   // identical for all waves in the block
    const int tN = wave % tilesN;
    const int n16  = lane & 15;
    const int half = lane >> 4;
    const int mBase = tM * 64;
    const int nBase = tN * 64;

    // staging role for the non-TDM path: 256 threads x 16B = 64x32 bf16 tile
    const int sRow = tid >> 2;   // 0..63
    const int sChk = tid & 3;    // 16-byte chunk within row

    auto stage = [&](int buf, int k0) {
#if HAVE_TDM
        if ((tid >> 5) == 0) {  // one TDM descriptor per block covers the tile
            const uint64_t gaddr =
                (uint64_t)(uintptr_t)A +
                ((uint64_t)mBase * (uint64_t)Kd + (uint64_t)k0) * 2u;
            const uint32_t laddr =
                (uint32_t)(uintptr_t)(AS3 void*)&ldsA[buf][0];
            u32x4v g0;
            g0[0] = 1u;                 // count=1, user descriptor, no gather
            g0[1] = laddr;              // LDS byte address
            g0[2] = (uint32_t)gaddr;    // global_addr[31:0]
            g0[3] = (uint32_t)(gaddr >> 32) | (2u << 30);  // addr[56:32]|type=2
            i32x8v g1;
            g1[0] = (int)((1u << 16)    // data_size = 2 bytes
                        | (1u << 20)    // pad_enable (LDS row padding)
                        | (3u << 22)    // pad_interval: every 64 bytes
                        | (3u << 25));  // pad_amount: 4 dwords
            g1[1] = (int)(32u << 16);   // tensor_dim0 = 32 (lo16 @ bits 63:48)
            g1[2] = (int)(64u << 16);   // dim0 hi16=0 | tensor_dim1 = 64 lo16
            g1[3] = (int)(32u << 16);   // dim1 hi16=0 | tile_dim0 = 32
            g1[4] = (int)(64u);         // tile_dim1 = 64 | tile_dim2 = 0
            g1[5] = (int)(uint32_t)Kd;  // tensor_dim0_stride lo32 (elements)
            g1[6] = 0;                  // stride0 hi16 | stride1 lo16
            g1[7] = 0;                  // stride1 hi32
            i32x4s g2 = {0, 0, 0, 0};   // 2D tensor: groups 2/3 unused
            i32x4s g3 = {0, 0, 0, 0};
            i32x8v g4 = {0, 0, 0, 0, 0, 0, 0, 0};  // 6-arg variant extra group
            __builtin_amdgcn_tensor_load_to_lds(g0, g1, g2, g3, g4, 0);
        }
#else
        const uint16_t* gp = A + (size_t)(mBase + sRow) * Kd + k0 + sChk * 8;
        uint32_t* lp = &ldsA[buf][sRow * LDS_ROW + sChk * 4];
#  if HAVE_ASYNC_LDS
        __builtin_amdgcn_global_load_async_to_lds_b128((AS1 i32x4v*)gp,
                                                       (AS3 i32x4v*)lp, 0, 0);
#  else
        *(u32x4*)lp = *(const u32x4*)gp;
#  endif
#endif
    };

    v8f c[4][4];
#pragma unroll
    for (int i = 0; i < 4; ++i)
#pragma unroll
        for (int j = 0; j < 4; ++j) c[i][j] = vzero8();

    stage(0, 0);
    int cur = 0;
    for (int k0 = 0; k0 < Kd; k0 += 32) {
        stage_wait();
        __syncthreads();
        if (k0 + 32 < Kd) stage(cur ^ 1, k0 + 32);  // overlap next DMA

        Frag b[4];
#pragma unroll
        for (int ni = 0; ni < 4; ++ni) {
            // B-matrix 32x16: lane(n,half) holds 16 contiguous K elems @16*half
            const uint32_t* p =
                (const uint32_t*)(W + (size_t)(nBase + ni * 16 + n16) * Kd + k0 +
                                  16 * half);
            b[ni].q[0] = ((const u32x4*)p)[0];
            b[ni].q[1] = ((const u32x4*)p)[1];
        }
        Frag a[4];
#pragma unroll
        for (int mi = 0; mi < 4; ++mi) {
            // A-matrix 16x32: lane(m,half) holds 16 contiguous K elems @16*half
            const uint32_t* lrow = &ldsA[cur][(mi * 16 + n16) * LDS_ROW];
            a[mi].q[0] = ((const u32x4*)lrow)[2 * half];
            a[mi].q[1] = ((const u32x4*)lrow)[2 * half + 1];
        }
#pragma unroll
        for (int mi = 0; mi < 4; ++mi)
#pragma unroll
            for (int ni = 0; ni < 4; ++ni)
                c[mi][ni] = wmma_bf16(a[mi].v, b[ni].v, c[mi][ni]);

        __syncthreads();
        cur ^= 1;
    }

    // store: C layout lane(n,half), vgpr v -> element (m = v+8*half, n)
#pragma unroll
    for (int mi = 0; mi < 4; ++mi)
#pragma unroll
        for (int ni = 0; ni < 4; ++ni)
#pragma unroll
            for (int v = 0; v < 8; ++v) {
                const int row = mBase + mi * 16 + v + 8 * half;
                const int col = nBase + ni * 16 + n16;
                const float val = c[mi][ni][v];
                if (OUT_MODE == 1) {
                    ((float*)Cout)[(size_t)row * N + col] = val;
                } else if (OUT_MODE == 0) {
                    ((uint16_t*)Cout)[(size_t)row * N + col] =
                        (uint16_t)f32_to_bf16_bits(val * outScale);
                } else {  // transposed per-head: Vt[b][h][d][s]
                    const int h  = col >> 6;
                    const int dl = col & (DKH - 1);
                    const int bb = row / SEQ;
                    const int ss = row - bb * SEQ;
                    ((uint16_t*)Cout)[(((size_t)(bb * HEADS + h) * DKH + dl) *
                                       SEQ) + ss] = (uint16_t)f32_to_bf16_bits(val);
                }
            }
}

// ---------------------------------------------------------------------------
// Flash attention step: 32 keys x 16 queries. Computes S^T = K·Q^T so softmax
// rows are per-lane C-fragment columns; the causal mask is compiled in only
// for the 1-2 tail steps (MASKED=true). Q is pre-scaled by 1/sqrt(dk).
// ---------------------------------------------------------------------------
template <bool MASKED>
__device__ __forceinline__ void flash_step(
    const uint16_t* __restrict__ Kb, const uint16_t* __restrict__ Vt,
    int kb, int b, int h, int qBase, int n16, int half,
    const Frag (&qf)[2], v8f (&ot)[4], float& m_run, float& l_run) {
    // ---- S^T chunk (two 16x16 C fragments)
    v8f s0 = vzero8(), s1 = vzero8();
    {
        const uint16_t* krow0 =
            Kb + (size_t)(b * SEQ + kb + n16) * D_MODEL + h * DKH;
        const uint16_t* krow1 = krow0 + 16 * D_MODEL;
        __builtin_prefetch(krow0 + 32 * D_MODEL, 0, 1);
#pragma unroll
        for (int dc = 0; dc < 2; ++dc) {
            Frag ka, kc;
            const uint32_t* p0 = (const uint32_t*)(krow0 + dc * 32);
            const uint32_t* p1 = (const uint32_t*)(krow1 + dc * 32);
            ka.q[0] = ((const u32x4*)p0)[2 * half];
            ka.q[1] = ((const u32x4*)p0)[2 * half + 1];
            kc.q[0] = ((const u32x4*)p1)[2 * half];
            kc.q[1] = ((const u32x4*)p1)[2 * half + 1];
            s0 = wmma_bf16(ka.v, qf[dc].v, s0);
            s1 = wmma_bf16(kc.v, qf[dc].v, s1);
        }
    }

    // ---- per-query (per-lane-column) online softmax
    const int qg = qBase + n16;
    float sv0[8], sv1[8];
    float mloc = -1e30f;
#pragma unroll
    for (int v = 0; v < 8; ++v) {
        float a = s0[v], cc = s1[v];
        if (MASKED) {
            const int kg0 = kb + v + 8 * half;
            a  = (kg0 <= qg) ? a : -1e30f;
            cc = (kg0 + 16 <= qg) ? cc : -1e30f;
        }
        sv0[v] = a;
        sv1[v] = cc;
        mloc = fmaxf(mloc, fmaxf(a, cc));
    }
    mloc = fmaxf(mloc, __shfl_xor(mloc, 16, 32));
    const float m_new = fmaxf(m_run, mloc);
    const float alpha = __expf(m_run - m_new);

    float p0[8], p1[8], psum = 0.f;
#pragma unroll
    for (int v = 0; v < 8; ++v) {
        p0[v] = __expf(sv0[v] - m_new);
        p1[v] = __expf(sv1[v] - m_new);
        psum += p0[v] + p1[v];
    }
    psum += __shfl_xor(psum, 16, 32);
    l_run = l_run * alpha + psum;
    m_run = m_new;

    // ---- build 32x16 bf16 B-fragment of P from the two C fragments
    float o0[8], o1[8];
#pragma unroll
    for (int v = 0; v < 8; ++v) {
        o0[v] = __shfl_xor(p0[v], 16, 32);
        o1[v] = __shfl_xor(p1[v], 16, 32);
    }
    float col[16];
#pragma unroll
    for (int k = 0; k < 8; ++k) {
        col[k]     = (half == 0) ? p0[k] : o1[k];
        col[k + 8] = (half == 0) ? o0[k] : p1[k];
    }
    Frag pf;
#pragma unroll
    for (int k = 0; k < 8; ++k) pf.u[k] = pack_bf16x2(col[2 * k], col[2 * k + 1]);

    // ---- rescale O^T (per-lane scalar) and accumulate V^T·P
#pragma unroll
    for (int ds = 0; ds < 4; ++ds)
#pragma unroll
        for (int v = 0; v < 8; ++v) ot[ds][v] *= alpha;

#pragma unroll
    for (int ds = 0; ds < 4; ++ds) {
        Frag va;  // A = V^T: row m = local d, K = keys (contiguous in Vt)
        const uint32_t* p =
            (const uint32_t*)(Vt +
                              ((size_t)((b * HEADS + h) * DKH + ds * 16 + n16)) *
                                  SEQ +
                              kb);
        va.q[0] = ((const u32x4*)p)[2 * half];
        va.q[1] = ((const u32x4*)p)[2 * half + 1];
        ot[ds] = wmma_bf16(va.v, pf.v, ot[ds]);
    }
}

__global__ __launch_bounds__(256)
void flash_attn_wmma(const uint16_t* __restrict__ Qb,   // [B*S, D] bf16 (pre-scaled)
                     const uint16_t* __restrict__ Kb,   // [B*S, D] bf16
                     const uint16_t* __restrict__ Vt,   // [B,H,64,S] bf16
                     uint16_t* __restrict__ Ob) {       // [B*S, D] bf16
    const int lane = threadIdx.x & 31;
    const int wave = blockIdx.x * (blockDim.x >> 5) + (threadIdx.x >> 5);
    const int qTiles = SEQ / 16;
    const int b   = wave / (HEADS * qTiles);
    const int rem = wave % (HEADS * qTiles);
    const int h   = rem / qTiles;
    const int qBase = (rem % qTiles) * 16;
    const int n16  = lane & 15;
    const int half = lane >> 4;

    // Q B-fragments (reused for the whole key loop): two 32-deep d chunks
    Frag qf[2];
    {
        const uint16_t* qrow =
            Qb + (size_t)(b * SEQ + qBase + n16) * D_MODEL + h * DKH;
#pragma unroll
        for (int dc = 0; dc < 2; ++dc) {
            const uint32_t* p = (const uint32_t*)(qrow + dc * 32 + 16 * half);
            qf[dc].q[0] = ((const u32x4*)p)[0];
            qf[dc].q[1] = ((const u32x4*)p)[1];
        }
    }

    v8f ot[4];
#pragma unroll
    for (int i = 0; i < 4; ++i) ot[i] = vzero8();
    float m_run = -1e30f, l_run = 0.f;

    const int qHi = qBase + 16;
    // steps with kb+31 <= qBase never hit the causal boundary
    const int kNoMask = (qBase >= 31) ? (((qBase - 31) / 32) * 32 + 32) : 0;

    int kb = 0;
    for (; kb < kNoMask; kb += 32)
        flash_step<false>(Kb, Vt, kb, b, h, qBase, n16, half, qf, ot, m_run,
                          l_run);
    for (; kb < qHi; kb += 32)
        flash_step<true>(Kb, Vt, kb, b, h, qBase, n16, half, qf, ot, m_run,
                         l_run);

    // ---- finalize: O[q, d] = O^T[d, q] / l ; concat heads into [B*S, D]
    const float inv_l = 1.0f / l_run;
    const size_t orow = (size_t)(b * SEQ + qBase + n16) * D_MODEL + h * DKH;
#pragma unroll
    for (int ds = 0; ds < 4; ++ds)
#pragma unroll
        for (int v = 0; v < 8; ++v)
            Ob[orow + ds * 16 + v + 8 * half] =
                (uint16_t)f32_to_bf16_bits(ot[ds][v] * inv_l);
}

// ---------------------------------------------------------------------------
extern "C" void kernel_launch(void* const* d_in, const int* in_sizes, int n_in,
                              void* d_out, int out_size, void* d_ws,
                              size_t ws_size, hipStream_t stream) {
    const float* Q  = (const float*)d_in[0];
    const float* K  = (const float*)d_in[1];
    const float* V  = (const float*)d_in[2];
    // d_in[3] = mask (causal, reconstructed analytically in-kernel)
    const float* Wq = (const float*)d_in[4];
    const float* Wk = (const float*)d_in[5];
    const float* Wv = (const float*)d_in[6];
    const float* Wo = (const float*)d_in[7];

    const size_t NTOK = (size_t)BATCH * SEQ;        // 4096 rows
    const size_t XE   = NTOK * D_MODEL;             // activation elements
    const size_t WE   = (size_t)D_MODEL * D_MODEL;  // weight elements

    uint16_t* ws = (uint16_t*)d_ws;
    uint16_t* xq = ws; ws += XE;
    uint16_t* xk = ws; ws += XE;
    uint16_t* xv = ws; ws += XE;
    uint16_t* wq = ws; ws += WE;
    uint16_t* wk = ws; ws += WE;
    uint16_t* wv = ws; ws += WE;
    uint16_t* wo = ws; ws += WE;
    uint16_t* qb = ws; ws += XE;
    uint16_t* kb = ws; ws += XE;
    uint16_t* vt = ws; ws += XE;   // [B,H,64,S]
    uint16_t* ao = ws; ws += XE;

    auto conv = [&](const float* s, uint16_t* d, size_t n) {
        f32_to_bf16_kernel<<<dim3((unsigned)((n + 255) / 256)), dim3(256), 0,
                             stream>>>(s, d, (int)n);
    };
    conv(Q, xq, XE);
    conv(K, xk, XE);
    conv(V, xv, XE);
    conv(Wq, wq, WE);
    conv(Wk, wk, WE);
    conv(Wv, wv, WE);
    conv(Wo, wo, WE);

    const int gemmWaves = (int)((NTOK / 64) * (D_MODEL / 64));  // 1024
    const dim3 gblocks(gemmWaves / 8), tpb(256);

    // fold 1/sqrt(d_k)=0.125 (exact in bf16) into the Q projection
    gemm_wmma_bf16<0><<<gblocks, tpb, 0, stream>>>(xq, wq, qb, (int)NTOK,
                                                   D_MODEL, D_MODEL, 0.125f);
    gemm_wmma_bf16<0><<<gblocks, tpb, 0, stream>>>(xk, wk, kb, (int)NTOK,
                                                   D_MODEL, D_MODEL, 1.0f);
    gemm_wmma_bf16<2><<<gblocks, tpb, 0, stream>>>(xv, wv, vt, (int)NTOK,
                                                   D_MODEL, D_MODEL, 1.0f);

    const int fwaves = BATCH * HEADS * (SEQ / 16);  // 4096
    flash_attn_wmma<<<dim3(fwaves / 8), tpb, 0, stream>>>(qb, kb, vt, ao);

    gemm_wmma_bf16<1><<<gblocks, tpb, 0, stream>>>(ao, wo, d_out, (int)NTOK,
                                                   D_MODEL, D_MODEL, 1.0f);
}
